// MLPPredictor_21174188769662
// MI455X (gfx1250) — matrix operand order repaired and checked by reading, not compiled, not run on hardware
//
#include <hip/hip_runtime.h>

typedef __attribute__((ext_vector_type(16))) _Float16 v16h;
typedef __attribute__((ext_vector_type(8)))  _Float16 v8h;
typedef __attribute__((ext_vector_type(4)))  _Float16 v4h;
typedef __attribute__((ext_vector_type(8)))  float    v8f;

union V16 {
    v16h v;
    struct { v8h lo; v8h hi; } p;
};

// ---------------------------------------------------------------------------
// f32 -> f16 conversion (vectorized x4): node_emb table and W1.
// ---------------------------------------------------------------------------
__global__ void f32_to_f16_kernel(const float* __restrict__ in,
                                  _Float16* __restrict__ out, int n4) {
    int i = blockIdx.x * blockDim.x + threadIdx.x;
    if (i >= n4) return;
    float4 v = ((const float4*)in)[i];
    v4h o;
    o[0] = (_Float16)v.x;
    o[1] = (_Float16)v.y;
    o[2] = (_Float16)v.z;
    o[3] = (_Float16)v.w;
    *(v4h*)(out + 4 * (size_t)i) = o;
}

// ---------------------------------------------------------------------------
// Edge MLP. Per wave, per "tile-pair": 32 edges (two M=16 tiles sharing every
// B fragment) x 128 hidden (8 N-tiles), K=256 in chunks of 32 -> 128 WMMAs.
// Grid-stride over tile-pairs; an asm memory clobber per iteration prevents
// LICM from hoisting the 64 LDS B-fragments into (spilled) registers.
// ---------------------------------------------------------------------------
__global__ __launch_bounds__(256)
void edge_mlp_kernel(const _Float16* __restrict__ node_f16,
                     const _Float16* __restrict__ w1_f16,
                     const float* __restrict__ b1,
                     const float* __restrict__ w2,
                     const float* __restrict__ b2,
                     const int* __restrict__ src,
                     const int* __restrict__ dst,
                     float* __restrict__ out,
                     int n_edges, int ntp) {
    __shared__ __align__(16) _Float16 ldsW1[128 * 256];  // 64 KB
    __shared__ float ldsB1[128];
    __shared__ float ldsW2[128];

    // One-time staging per block.
    {
        const v8h* g = (const v8h*)w1_f16;
        v8h* l = (v8h*)ldsW1;
        for (int i = threadIdx.x; i < (128 * 256) / 8; i += blockDim.x)
            l[i] = g[i];
        if (threadIdx.x < 128) {
            ldsB1[threadIdx.x] = b1[threadIdx.x];
            ldsW2[threadIdx.x] = w2[threadIdx.x];
        }
    }
    __syncthreads();

    const int wave = threadIdx.x >> 5;
    const int lane = threadIdx.x & 31;
    const int row  = lane & 15;        // A row / B column within a tile
    const int hi   = (lane >> 4) & 1;  // half-wave selector per ISA layout
    const float bias2 = b2[0];

#pragma unroll 1
    for (int tp = blockIdx.x * 8 + wave; tp < ntp; tp += gridDim.x * 8) {
        // Opaque memory clobber: stops LICM from hoisting the loop-invariant
        // LDS B-fragment loads out of this loop (which caused 512-VGPR
        // pressure and scratch spills). B must be re-read from LDS each pass.
        asm volatile("" ::: "memory");

        const int e0 = tp * 32;

        // Gather endpoint indices for this lane's two edges (clamped tail).
        int ea = e0 + row;        if (ea >= n_edges) ea = n_edges - 1;
        int eb = e0 + 16 + row;   if (eb >= n_edges) eb = n_edges - 1;
        const int sa = src[ea], da = dst[ea];
        const int sb = src[eb], db = dst[eb];

        v8f acc0[8], acc1[8];
#pragma unroll
        for (int nt = 0; nt < 8; ++nt) {
            acc0[nt] = (v8f)0.0f;
            acc1[nt] = (v8f)0.0f;
        }

        // K = 256 = concat(src 0..127, dst 128..255), chunks of 32.
#pragma unroll
        for (int kc = 0; kc < 8; ++kc) {
            const int na = (kc < 4) ? sa : da;
            const int nb = (kc < 4) ? sb : db;
            const int kin = (kc & 3) * 32 + hi * 8;

            V16 a0, a1;
            const _Float16* pa = node_f16 + (size_t)na * 128 + kin;
            const _Float16* pb = node_f16 + (size_t)nb * 128 + kin;
            a0.p.lo = *(const v8h*)(pa);
            a0.p.hi = *(const v8h*)(pa + 16);
            a1.p.lo = *(const v8h*)(pb);
            a1.p.hi = *(const v8h*)(pb + 16);

#pragma unroll
            for (int nt = 0; nt < 8; ++nt) {
                const _Float16* bb =
                    &ldsW1[(nt * 16 + row) * 256 + kc * 32 + hi * 8];
                V16 bf;
                bf.p.lo = *(const v8h*)(bb);
                bf.p.hi = *(const v8h*)(bb + 16);

                // Two independent WMMAs share one B fragment: halves LDS
                // traffic per FLOP and breaks the ds-wait -> wmma chain.
                acc0[nt] = __builtin_amdgcn_wmma_f32_16x16x32_f16(
                    false, a0.v, false, bf.v, (short)0, acc0[nt], false, false);
                acc1[nt] = __builtin_amdgcn_wmma_f32_16x16x32_f16(
                    false, a1.v, false, bf.v, (short)0, acc1[nt], false, false);
            }
        }

        // Epilogue: relu(acc + b1[n]) * W2[n], reduce over N.
        // C/D layout: VGPR j holds M=j (lanes 0-15) / M=j+8 (lanes 16-31).
        float p0[8], p1[8];
#pragma unroll
        for (int j = 0; j < 8; ++j) { p0[j] = 0.0f; p1[j] = 0.0f; }

#pragma unroll
        for (int nt = 0; nt < 8; ++nt) {
            const int n = nt * 16 + row;
            const float b1v = ldsB1[n];
            const float w2v = ldsW2[n];
#pragma unroll
            for (int j = 0; j < 8; ++j) {
                float h0 = acc0[nt][j] + b1v;
                float h1 = acc1[nt][j] + b1v;
                h0 = h0 > 0.0f ? h0 : 0.0f;
                h1 = h1 > 0.0f ? h1 : 0.0f;
                p0[j] += h0 * w2v;
                p1[j] += h1 * w2v;
            }
        }

        // Reduce across the 16 N-lanes of each half-wave.
#pragma unroll
        for (int m = 1; m < 16; m <<= 1) {
#pragma unroll
            for (int j = 0; j < 8; ++j) {
                p0[j] += __shfl_xor(p0[j], m, 32);
                p1[j] += __shfl_xor(p1[j], m, 32);
            }
        }

        if (row == 0) {
            const int mb0 = e0 + hi * 8;        // tile 0: lanes 0/16
            const int mb1 = e0 + 16 + hi * 8;   // tile 1
#pragma unroll
            for (int j = 0; j < 8; ++j) {
                if (mb0 + j < n_edges) out[mb0 + j] = p0[j] + bias2;
                if (mb1 + j < n_edges) out[mb1 + j] = p1[j] + bias2;
            }
        }
    }
}

// ---------------------------------------------------------------------------
// Inputs (setup_inputs order): node_emb, W1, b1, W2, b2, src, dst
// ---------------------------------------------------------------------------
extern "C" void kernel_launch(void* const* d_in, const int* in_sizes, int n_in,
                              void* d_out, int out_size, void* d_ws,
                              size_t ws_size, hipStream_t stream) {
    const float* node_emb = (const float*)d_in[0];
    const float* W1       = (const float*)d_in[1];
    const float* b1       = (const float*)d_in[2];
    const float* W2       = (const float*)d_in[3];
    const float* b2       = (const float*)d_in[4];
    const int*   src      = (const int*)d_in[5];
    const int*   dst      = (const int*)d_in[6];
    float*       out      = (float*)d_out;

    const int node_elems = in_sizes[0];   // 50000 * 128
    const int w1_elems   = in_sizes[1];   // 128 * 256
    const int n_edges    = in_sizes[5];   // 600000

    // Workspace: f16 node table (12.8 MB) followed by f16 W1 (64 KB).
    _Float16* node_f16 = (_Float16*)d_ws;
    _Float16* w1_f16   = (_Float16*)((char*)d_ws + (size_t)node_elems * 2);

    const int n4_nodes = node_elems / 4;
    f32_to_f16_kernel<<<(n4_nodes + 255) / 256, 256, 0, stream>>>(
        node_emb, node_f16, n4_nodes);

    const int n4_w1 = w1_elems / 4;
    f32_to_f16_kernel<<<(n4_w1 + 255) / 256, 256, 0, stream>>>(
        W1, w1_f16, n4_w1);

    const int ntiles = (n_edges + 15) / 16;  // 37500
    const int ntp    = (ntiles + 1) / 2;     // 18750 tile-pairs (32 edges each)
    int nblocks = (ntp + 7) / 8;
    if (nblocks > 1024) nblocks = 1024;      // amortize 64 KB LDS staging
    edge_mlp_kernel<<<nblocks, 256, 0, stream>>>(
        node_f16, w1_f16, b1, W2, b2, src, dst, out, n_edges, ntp);
}
